// Enhanced_transformer_52725018526347
// MI455X (gfx1250) — matrix operand order repaired and checked by reading, not compiled, not run on hardware
//
#include <hip/hip_runtime.h>
#include <hip/hip_bf16.h>

// ---------------------------------------------------------------------------
// Types
// ---------------------------------------------------------------------------
typedef __bf16 bf16_t;
typedef __attribute__((ext_vector_type(16))) __bf16 bf16x16;
typedef __attribute__((ext_vector_type(8)))  __bf16 bf16x8;
typedef __attribute__((ext_vector_type(4)))  __bf16 bf16x4;
typedef __attribute__((ext_vector_type(8)))  float  v8f;

#define B_SZ 8
#define N_SZ 4096
#define P_SZ 1024
#define P4_SZ 256

__device__ __forceinline__ bf16_t f2bf(float f) { return (bf16_t)f; }  // native v_cvt

__device__ __forceinline__ float gelu_exact(float x) {
    return 0.5f * x * (1.0f + erff(x * 0.70710678118654752f));
}

// CDNA5 async global->LDS fill (ASYNCcnt path).  On gfx1250 the low 32 bits of
// a flat shared-memory address are the LDS offset (ISA 10.2 aperture rules).
__device__ __forceinline__ void async_ld_b128(void* lds_ptr, unsigned goff_bytes,
                                              const void* sbase) {
    unsigned lds = (unsigned)(unsigned long long)lds_ptr;
    asm volatile("global_load_async_to_lds_b128 %0, %1, %2 offset:0"
                 :: "v"(lds), "v"(goff_bytes), "s"(sbase) : "memory");
}
__device__ __forceinline__ void wait_async0() {
    asm volatile("s_wait_asynccnt 0" ::: "memory");
}

// ---------------------------------------------------------------------------
// Generic bf16 WMMA GEMM: C = act(A*B + bias) [+ Res]
//   TRANSA: A[m,k] = Ag[k*lda + m]  (else Ag[m*lda + k])
//   TRANSB: B[k,n] = Bg[n*ldb + k]  (else Bg[k*ldb + n])
//   BIAS  : 0 none, 1 per-N column, 2 per-M row
//   GELU  : 0/1 (applied after bias)
//   STORE : 0 bf16 out, 1 f32 out, 2 f32 out = val + Res
// Workgroup tile 128x128 (8 waves, 32x64 per wave), BK=32,
// double-buffered LDS, async global->LDS fills on non-transposed operands.
// ---------------------------------------------------------------------------
template<int TRANSA, int TRANSB, int BIAS, int GELU, int STORE>
__global__ __launch_bounds__(256)
void gemm_wmma_bf16(const bf16_t* __restrict__ Ag, long long sA, int lda,
                    const bf16_t* __restrict__ Bg, long long sB, int ldb,
                    const float*  __restrict__ bias,
                    const float*  __restrict__ Res, long long sR,
                    void* __restrict__ Out, long long sO, int ldc,
                    int M, int N, int K)
{
    constexpr int BM = 128, BN = 128, BK = 32;
    constexpr int LAS = BK + 8;    // 40 bf16 (80B rows, 16B aligned)
    constexpr int LBS = BN + 8;    // 136 bf16 (272B rows, 16B aligned)
    __shared__ bf16_t As[2][BM][LAS];
    __shared__ bf16_t Bs[2][BK][LBS];

    const int bz = blockIdx.z;
    const int m0 = blockIdx.y * BM;
    const int n0 = blockIdx.x * BN;
    const int t  = threadIdx.x;
    const int w  = t >> 5;
    const int l  = t & 31;
    const int wm = (w & 3) * 32;    // 4 wave rows   -> 128
    const int wn = (w >> 2) * 64;   // 2 wave cols   -> 128
    const int half  = l >> 4;
    const int row16 = l & 15;

    const bf16_t* A = Ag + (long long)bz * sA;
    const bf16_t* B = Bg + (long long)bz * sB;
    const int nk = K / BK;

    v8f acc[2][4];
#pragma unroll
    for (int i = 0; i < 2; ++i)
#pragma unroll
        for (int j = 0; j < 4; ++j)
#pragma unroll
            for (int r = 0; r < 8; ++r) acc[i][j][r] = 0.0f;

    bf16x8 ra[2], rb[2];

    // ---- staging helpers (branches fold at compile time) ----
    auto stage_async = [&](int k0, int buf) {
        if (TRANSA == 0) {
#pragma unroll
            for (int p = 0; p < 2; ++p) {
                int chunk = t + p * 256;
                int r = chunk >> 2;            // 0..127 (m)
                int c = (chunk & 3) * 8;       // 0..24  (k)
                async_ld_b128(&As[buf][r][c],
                              (unsigned)(((long long)(m0 + r) * lda + k0 + c) * 2), A);
            }
        }
        if (TRANSB == 0) {
#pragma unroll
            for (int p = 0; p < 2; ++p) {
                int chunk = t + p * 256;
                int r = chunk >> 4;            // 0..31  (k)
                int c = (chunk & 15) * 8;      // 0..120 (n)
                async_ld_b128(&Bs[buf][r][c],
                              (unsigned)(((long long)(k0 + r) * ldb + n0 + c) * 2), B);
            }
        }
    };
    auto load_regs = [&](int k0) {
        if (TRANSA == 1) {
#pragma unroll
            for (int p = 0; p < 2; ++p) {
                int chunk = t + p * 256;
                int r = chunk >> 4;            // k 0..31
                int c = (chunk & 15) * 8;      // m 0..120
                ra[p] = *(const bf16x8*)&A[(long long)(k0 + r) * lda + m0 + c];
            }
        }
        if (TRANSB == 1) {
#pragma unroll
            for (int p = 0; p < 2; ++p) {
                int chunk = t + p * 256;
                int r = chunk >> 2;            // n 0..127
                int c = (chunk & 3) * 8;       // k 0..24
                rb[p] = *(const bf16x8*)&B[(long long)(n0 + r) * ldb + k0 + c];
            }
        }
    };
    auto store_regs = [&](int buf) {
        if (TRANSA == 1) {
#pragma unroll
            for (int p = 0; p < 2; ++p) {
                int chunk = t + p * 256;
                int r = chunk >> 4;
                int c = (chunk & 15) * 8;
#pragma unroll
                for (int j = 0; j < 8; ++j) As[buf][c + j][r] = ra[p][j];
            }
        }
        if (TRANSB == 1) {
#pragma unroll
            for (int p = 0; p < 2; ++p) {
                int chunk = t + p * 256;
                int r = chunk >> 2;
                int c = (chunk & 3) * 8;
#pragma unroll
                for (int j = 0; j < 8; ++j) Bs[buf][c + j][r] = rb[p][j];
            }
        }
    };

    // ---- prologue: fill buffer 0 ----
    stage_async(0, 0);
    load_regs(0);
    store_regs(0);

    for (int kt = 0; kt < nk; ++kt) {
        wait_async0();
        __syncthreads();
        const int cb = kt & 1;
        const int nb = cb ^ 1;

        if (kt + 1 < nk) {
            stage_async((kt + 1) * BK, nb);
            load_regs((kt + 1) * BK);
        }
        if (kt + 2 < nk) {   // L2 pre-warm two steps ahead -> global_prefetch_b8
            int k0n = (kt + 2) * BK;
            __builtin_prefetch((const void*)&A[(long long)(m0 + (t >> 1)) * lda + k0n], 0, 0);
            __builtin_prefetch((const void*)&B[(long long)(k0n + (t >> 3)) * ldb + n0], 0, 0);
        }

        // ---- fragments per CDNA5 16-bit WMMA VGPR layouts ----
        bf16x16 af[2], bf[4];
#pragma unroll
        for (int i = 0; i < 2; ++i) {
            const bf16_t* ap = &As[cb][wm + i * 16 + row16][half * 8];
#pragma unroll
            for (int j = 0; j < 8; ++j) {
                af[i][j]     = ap[j];
                af[i][j + 8] = ap[16 + j];
            }
        }
#pragma unroll
        for (int j = 0; j < 4; ++j) {
            const bf16_t* bp = &Bs[cb][half * 8][wn + j * 16 + row16];
#pragma unroll
            for (int q = 0; q < 8; ++q) {
                bf[j][q]     = bp[q * LBS];
                bf[j][q + 8] = bp[(16 + q) * LBS];
            }
        }
#pragma unroll
        for (int i = 0; i < 2; ++i)
#pragma unroll
            for (int j = 0; j < 4; ++j)
                acc[i][j] = __builtin_amdgcn_wmma_f32_16x16x32_bf16(
                    false, af[i], false, bf[j], (short)0, acc[i][j], false, false);

        if (kt + 1 < nk) store_regs(nb);
    }

    // ---- epilogue: bias / gelu / residual / store ----
#pragma unroll
    for (int i = 0; i < 2; ++i) {
#pragma unroll
        for (int j = 0; j < 4; ++j) {
#pragma unroll
            for (int r = 0; r < 8; ++r) {
                int gm = m0 + wm + i * 16 + half * 8 + r;
                int gn = n0 + wn + j * 16 + row16;
                float v = acc[i][j][r];
                if (BIAS == 1) v += bias[gn];
                if (BIAS == 2) v += bias[gm];
                if (GELU)      v = gelu_exact(v);
                long long idx = (long long)gm * ldc + gn;
                if (STORE == 0) {
                    ((bf16_t*)Out)[(long long)bz * sO + idx] = f2bf(v);
                } else if (STORE == 1) {
                    ((float*)Out)[(long long)bz * sO + idx] = v;
                } else {
                    ((float*)Out)[(long long)bz * sO + idx] =
                        v + Res[(long long)bz * sR + idx];
                }
            }
        }
    }
}

// ---------------------------------------------------------------------------
// LayerNorm over rows of 1024 (fp32 in -> bf16 out), one block per row.
// ---------------------------------------------------------------------------
__global__ __launch_bounds__(256)
void ln_rows_kernel(const float* __restrict__ X, const float* __restrict__ gam,
                    const float* __restrict__ bet, bf16_t* __restrict__ H)
{
    const long long row = blockIdx.x;
    const int t = threadIdx.x;
    const float* x = X + row * (long long)P_SZ;
    float4 v = *(const float4*)(x + t * 4);
    float s  = v.x + v.y + v.z + v.w;
    float ss = v.x * v.x + v.y * v.y + v.z * v.z + v.w * v.w;
#pragma unroll
    for (int o = 16; o > 0; o >>= 1) {
        s  += __shfl_xor(s,  o, 32);
        ss += __shfl_xor(ss, o, 32);
    }
    __shared__ float rs[8], rss[8];
    __shared__ float s_mu, s_ri;
    if ((t & 31) == 0) { rs[t >> 5] = s; rss[t >> 5] = ss; }
    __syncthreads();
    if (t == 0) {
        float a = 0.f, b = 0.f;
#pragma unroll
        for (int i = 0; i < 8; ++i) { a += rs[i]; b += rss[i]; }
        float mu  = a * (1.0f / P_SZ);
        float var = b * (1.0f / P_SZ) - mu * mu;
        s_mu = mu;
        s_ri = rsqrtf(var + 1e-5f);
    }
    __syncthreads();
    float mu = s_mu, ri = s_ri;
    float4 g4 = *(const float4*)(gam + t * 4);
    float4 b4 = *(const float4*)(bet + t * 4);
    bf16x4 o4;
    o4[0] = f2bf((v.x - mu) * ri * g4.x + b4.x);
    o4[1] = f2bf((v.y - mu) * ri * g4.y + b4.y);
    o4[2] = f2bf((v.z - mu) * ri * g4.z + b4.z);
    o4[3] = f2bf((v.w - mu) * ri * g4.w + b4.w);
    *(bf16x4*)(H + row * (long long)P_SZ + t * 4) = o4;
}

// ---------------------------------------------------------------------------
// Softmax over rows of 1024 (fp32 in -> bf16 out), one block per row.
// ---------------------------------------------------------------------------
__global__ __launch_bounds__(256)
void softmax_rows_kernel(const float* __restrict__ X, bf16_t* __restrict__ Y)
{
    const long long row = blockIdx.x;
    const int t = threadIdx.x;
    const float* x = X + row * (long long)P_SZ;
    float4 v = *(const float4*)(x + t * 4);
    float mx = fmaxf(fmaxf(v.x, v.y), fmaxf(v.z, v.w));
#pragma unroll
    for (int o = 16; o > 0; o >>= 1) mx = fmaxf(mx, __shfl_xor(mx, o, 32));
    __shared__ float rm[8], rsum[8];
    __shared__ float s_mx, s_inv;
    if ((t & 31) == 0) rm[t >> 5] = mx;
    __syncthreads();
    if (t == 0) {
        float m = rm[0];
#pragma unroll
        for (int i = 1; i < 8; ++i) m = fmaxf(m, rm[i]);
        s_mx = m;
    }
    __syncthreads();
    float m = s_mx;
    float e0 = __expf(v.x - m), e1 = __expf(v.y - m);
    float e2 = __expf(v.z - m), e3 = __expf(v.w - m);
    float s = e0 + e1 + e2 + e3;
#pragma unroll
    for (int o = 16; o > 0; o >>= 1) s += __shfl_xor(s, o, 32);
    if ((t & 31) == 0) rsum[t >> 5] = s;
    __syncthreads();
    if (t == 0) {
        float a = 0.f;
#pragma unroll
        for (int i = 0; i < 8; ++i) a += rsum[i];
        s_inv = 1.0f / a;
    }
    __syncthreads();
    float inv = s_inv;
    bf16x4 o4;
    o4[0] = f2bf(e0 * inv);
    o4[1] = f2bf(e1 * inv);
    o4[2] = f2bf(e2 * inv);
    o4[3] = f2bf(e3 * inv);
    *(bf16x4*)(Y + row * (long long)P_SZ + t * 4) = o4;
}

// ---------------------------------------------------------------------------
// fp32 -> bf16 conversion (for weights)
// ---------------------------------------------------------------------------
__global__ __launch_bounds__(256)
void cvt_f32_bf16_kernel(const float* __restrict__ src, bf16_t* __restrict__ dst, int n)
{
    int i = blockIdx.x * 256 + threadIdx.x;
    if (i < n) dst[i] = f2bf(src[i]);
}

// ---------------------------------------------------------------------------
// Launch
// ---------------------------------------------------------------------------
extern "C" void kernel_launch(void* const* d_in, const int* in_sizes, int n_in,
                              void* d_out, int out_size, void* d_ws, size_t ws_size,
                              hipStream_t stream)
{
    (void)in_sizes; (void)n_in; (void)out_size; (void)ws_size;

    // inputs (setup_inputs order)
    const float* x_in  = (const float*)d_in[1];
    const float* ln1_g = (const float*)d_in[2];
    const float* ln1_b = (const float*)d_in[3];
    const float* ln2_g = (const float*)d_in[4];
    const float* ln2_b = (const float*)d_in[5];
    const float* qk_w  = (const float*)d_in[6];
    const float* v_w   = (const float*)d_in[7];
    const float* v_b   = (const float*)d_in[8];
    const float* t1_w  = (const float*)d_in[9];
    const float* t1_b  = (const float*)d_in[10];
    const float* t2_w  = (const float*)d_in[11];
    const float* t2_b  = (const float*)d_in[12];
    const float* m1_w  = (const float*)d_in[13];
    const float* m1_b  = (const float*)d_in[14];
    const float* m2_w  = (const float*)d_in[15];
    const float* m2_b  = (const float*)d_in[16];
    float* out = (float*)d_out;

    const long long BN_ = (long long)B_SZ * N_SZ;       // 32768 rows
    const long long NP  = (long long)N_SZ * P_SZ;       // per-batch x stride
    const long long NQ  = (long long)N_SZ * P4_SZ;      // per-batch x_q stride
    const long long EE  = (long long)P4_SZ * P4_SZ;     // energy stride
    const long long A1S = (long long)P_SZ * P4_SZ;      // att1 stride
    const long long A2S = (long long)P_SZ * P_SZ;       // att2 stride

    // workspace layout
    char* base = (char*)d_ws;
    bf16_t* h    = (bf16_t*)(base + 0);                  // 64MB: h, later h2
    float*  a2   = (float*)(base + 0);                   // 32MB overlay (h dead)
    bf16_t* xv   = (bf16_t*)(base + 67108864);           // 64MB: x_v, later m
    bf16_t* m_   = xv;
    bf16_t* xq   = (bf16_t*)(base + 134217728);          // 16MB: x_q, later att_sm
    bf16_t* asm_ = xq;
    bf16_t* ene  = (bf16_t*)(base + 150994944);          // 1MB
    bf16_t* a1   = (bf16_t*)(base + 152043520);          // 4MB
    char*   wb   = base + 156237824;                     // bf16 weights (7.5MB)
    bf16_t* wqk = (bf16_t*)(wb + 0);
    bf16_t* wv  = (bf16_t*)(wb + 524288);
    bf16_t* wt1 = (bf16_t*)(wb + 2621440);
    bf16_t* wt2 = (bf16_t*)(wb + 3145728);
    bf16_t* wm1 = (bf16_t*)(wb + 3670016);
    bf16_t* wm2 = (bf16_t*)(wb + 5767168);

    // 1) weights -> bf16
    cvt_f32_bf16_kernel<<<(P4_SZ * P_SZ + 255) / 256, 256, 0, stream>>>(qk_w, wqk, P4_SZ * P_SZ);
    cvt_f32_bf16_kernel<<<(P_SZ  * P_SZ + 255) / 256, 256, 0, stream>>>(v_w,  wv,  P_SZ * P_SZ);
    cvt_f32_bf16_kernel<<<(P_SZ * P4_SZ + 255) / 256, 256, 0, stream>>>(t1_w, wt1, P_SZ * P4_SZ);
    cvt_f32_bf16_kernel<<<(P_SZ * P4_SZ + 255) / 256, 256, 0, stream>>>(t2_w, wt2, P_SZ * P4_SZ);
    cvt_f32_bf16_kernel<<<(P_SZ  * P_SZ + 255) / 256, 256, 0, stream>>>(m1_w, wm1, P_SZ * P_SZ);
    cvt_f32_bf16_kernel<<<(P_SZ  * P_SZ + 255) / 256, 256, 0, stream>>>(m2_w, wm2, P_SZ * P_SZ);

    // 2) h = LN1(x)  (bf16)
    ln_rows_kernel<<<(unsigned)BN_, 256, 0, stream>>>(x_in, ln1_g, ln1_b, h);

    // 3) x_q[n,o] = sum_p h[n,p] * qk_w[o,p]   (B folded into M)
    gemm_wmma_bf16<0, 1, 0, 0, 0><<<dim3(P4_SZ / 128, (unsigned)(BN_ / 128), 1), 256, 0, stream>>>(
        h, 0, P_SZ, wqk, 0, P_SZ, nullptr, nullptr, 0, xq, 0, P4_SZ,
        (int)BN_, P4_SZ, P_SZ);

    // 4) x_v = h @ v_w^T + v_b  (bf16)
    gemm_wmma_bf16<0, 1, 1, 0, 0><<<dim3(P_SZ / 128, (unsigned)(BN_ / 128), 1), 256, 0, stream>>>(
        h, 0, P_SZ, wv, 0, P_SZ, v_b, nullptr, 0, xv, 0, P_SZ,
        (int)BN_, P_SZ, P_SZ);

    // 5) energy[b] = x_q[b]^T @ x_q[b]  (256x256, K=4096) -> bf16
    gemm_wmma_bf16<1, 0, 0, 0, 0><<<dim3(P4_SZ / 128, P4_SZ / 128, B_SZ), 256, 0, stream>>>(
        xq, NQ, P4_SZ, xq, NQ, P4_SZ, nullptr, nullptr, 0, ene, EE, P4_SZ,
        P4_SZ, P4_SZ, N_SZ);

    // 6) att1[b] = gelu(t1_w @ energy[b] + t1_b[:,None])  -> bf16 [1024,256]
    gemm_wmma_bf16<0, 0, 2, 1, 0><<<dim3(P4_SZ / 128, P_SZ / 128, B_SZ), 256, 0, stream>>>(
        wt1, 0, P4_SZ, ene, EE, P4_SZ, t1_b, nullptr, 0, a1, A1S, P4_SZ,
        P_SZ, P4_SZ, P4_SZ);

    // 7) att2[b] = t2_w @ att1[b]^T + t2_b[:,None]  -> fp32 [1024,1024]
    gemm_wmma_bf16<0, 1, 2, 0, 1><<<dim3(P_SZ / 128, P_SZ / 128, B_SZ), 256, 0, stream>>>(
        wt2, 0, P4_SZ, a1, A1S, P4_SZ, t2_b, nullptr, 0, a2, A2S, P_SZ,
        P_SZ, P_SZ, P4_SZ);

    // 8) att_sm = softmax(att2, axis=-1) -> bf16
    softmax_rows_kernel<<<(unsigned)(B_SZ * P_SZ), 256, 0, stream>>>(a2, asm_);

    // 9) x1 = x_v[b] @ att_sm[b] + x  -> fp32 into d_out
    gemm_wmma_bf16<0, 0, 0, 0, 2><<<dim3(P_SZ / 128, N_SZ / 128, B_SZ), 256, 0, stream>>>(
        xv, NP, P_SZ, asm_, A2S, P_SZ, nullptr, x_in, NP, out, NP, P_SZ,
        N_SZ, P_SZ, P_SZ);

    // 10) h2 = LN2(x1)  (bf16, reuses h buffer)
    ln_rows_kernel<<<(unsigned)BN_, 256, 0, stream>>>(out, ln2_g, ln2_b, h);

    // 11) m = gelu(h2 @ m1_w^T + m1_b)  (bf16, reuses x_v buffer)
    gemm_wmma_bf16<0, 1, 1, 1, 0><<<dim3(P_SZ / 128, (unsigned)(BN_ / 128), 1), 256, 0, stream>>>(
        h, 0, P_SZ, wm1, 0, P_SZ, m1_b, nullptr, 0, m_, 0, P_SZ,
        (int)BN_, P_SZ, P_SZ);

    // 12) out = x1 + m @ m2_w^T + m2_b  (fused read-modify of d_out)
    gemm_wmma_bf16<0, 1, 1, 0, 2><<<dim3(P_SZ / 128, (unsigned)(BN_ / 128), 1), 256, 0, stream>>>(
        m_, 0, P_SZ, wm2, 0, P_SZ, m2_b, out, 0, out, 0, P_SZ,
        (int)BN_, P_SZ, P_SZ);
}
